// TSMambaBlock_19653770346660
// MI455X (gfx1250) — matrix operand order repaired and checked by reading, not compiled, not run on hardware
//
#include <hip/hip_runtime.h>
#include <hip/hip_bf16.h>

typedef __attribute__((ext_vector_type(16))) _Float16 v16h;
typedef __attribute__((ext_vector_type(8)))  float    v8f;

// ---------------------------------------------------------------- helpers
__device__ __forceinline__ float silu_f(float x)     { return x / (1.f + __expf(-x)); }
__device__ __forceinline__ float softplus_f(float x) { return (x > 20.f) ? x : log1pf(__expf(x)); }

__device__ __forceinline__ v8f wmma16(v16h a, v16h b, v8f c) {
  // D(16x16,f32) = A(16x32,f16) * B(32x16,f16) + C
  return __builtin_amdgcn_wmma_f32_16x16x32_f16(false, a, false, b, (short)0, c, false, false);
}

__device__ __forceinline__ v8f zero8() {
  v8f z;
#pragma unroll
  for (int r = 0; r < 8; ++r) z[r] = 0.f;
  return z;
}

// A-matrix fragment (16x32 f16): lane l row M=l%16;
// elems 0..7 -> K = 8*(l/16)+e ; elems 8..15 -> K = 16 + 8*(l/16) + (e-8)
__device__ __forceinline__ v16h load_a(const _Float16* src, int rowStride, int k0, int lane) {
  const int m = lane & 15, hl = lane >> 4;
  const _Float16* p = src + m * rowStride + k0 + 8 * hl;
  v16h a;
#pragma unroll
  for (int e = 0; e < 8; ++e) a[e] = p[e];
#pragma unroll
  for (int e = 0; e < 8; ++e) a[8 + e] = p[16 + e];
  return a;
}

// B-matrix fragment from pre-packed layout: 32 lanes x 16 halves contiguous per fragment
__device__ __forceinline__ v16h load_b_pk(const _Float16* packed, int fragIdx, int lane) {
  const _Float16* p = packed + ((size_t)fragIdx * 32 + lane) * 16;
  v16h b;
#pragma unroll
  for (int e = 0; e < 16; ++e) b[e] = p[e];
  return b;
}

// ------------------------------------------------------------- pack kernel
// B[k][n] fragment layout: frag(kt,nt) -> idx ((kt*NT+nt)*32 + lane)*16 + e
// with n = nt*16 + lane%16 ; k = kt*32 + 8*(lane/16) + (e%8) + 16*(e/8)
// transposeW=1: src is W(Nreal,Kreal), B[k][n]=W[n][k]  (for  X @ W^T)
// transposeW=0: src is W(Kreal,Nreal), B[k][n]=W[k][n]  (for  X @ W)
__global__ void pack_weights(const float* __restrict__ src, _Float16* __restrict__ dst,
                             int KT, int NT, int Kreal, int Nreal, int transposeW) {
  int idx = blockIdx.x * blockDim.x + threadIdx.x;
  int total = KT * NT * 512;
  if (idx >= total) return;
  int e    = idx & 15;
  int lane = (idx >> 4) & 31;
  int frag = idx >> 9;
  int nt = frag % NT, kt = frag / NT;
  int n = nt * 16 + (lane & 15);
  int k = kt * 32 + 8 * (lane >> 4) + (e & 7) + 16 * (e >> 3);
  float v = 0.f;
  if (n < Nreal && k < Kreal)
    v = transposeW ? src[n * Kreal + k] : src[k * Nreal + n];
  dst[idx] = (_Float16)v;
}

// ------------------------------------------------------- input transpose
// xt[f][t][c] = x[0][c][t][f]   (x: (1,64,256,64))
__global__ void transpose_in(const float* __restrict__ x, float* __restrict__ xt) {
  int i = blockIdx.x * blockDim.x + threadIdx.x;
  if (i >= 64 * 256 * 64) return;
  int c = i & 63, t = (i >> 6) & 255, f = i >> 14;
  xt[i] = x[(c * 256 + t) * 64 + f];
}

// ---------------------------------------------------------- block params
struct BlockParams {
  const float *norm_w, *norm2_w;
  const float *conv_w, *conv_b;
  const float *dt_w,  *dt_b;
  const float *A_log, *Dp;
  const _Float16 *w_in, *w_x, *w_out, *w_fc1, *w_fc2; // packed f16 fragments
};

// ------------------------------------------- fused Mamba block (one dir)
// One workgroup per sequence; 128 threads (4 waves); streams 16-row tiles.
// input element addr: seq*seqStride + row_in*rowStride + c  (row_in flipped for dir=1)
// output: cat[(seq*L + row_orig)*128 + dir*64 + c]  (f16)
__global__ void __launch_bounds__(128)
mamba_block_kernel(const float* __restrict__ xin, int L, int rowStride, int seqStride,
                   int dir, BlockParams p, _Float16* __restrict__ cat) {
  // s_r0 layout: [16][512] f16 ; GEMM outputs store at [row*512 + n0] always.
  //   in_proj:  cols 0..255 = xs, 256..511 = z
  //   fc1:      cols 0..255 = y1, 256..511 = gate
  __shared__ _Float16 s_r0[16 * 512];
  __shared__ float    s_xres[16 * 64];  // residual tile (f32), becomes residual2
  __shared__ _Float16 s_hn[16 * 64];    // normed activations (A frags)
  __shared__ _Float16 s_u[16 * 256];    // conv output u; reused as mlp hidden
  __shared__ float    s_xdbl[16 * 48];  // x_proj output (36 valid, padded to 48)
  __shared__ _Float16 s_y[16 * 256];    // scan output (gated)
  __shared__ _Float16 s_tail[3 * 256];  // conv history (last 3 xs rows), zero-init
  __shared__ float    s_nw[64], s_nw2[64];

  const int tid  = threadIdx.x;
  const int lane = tid & 31;
  const int wave = tid >> 5;
  const int seq  = blockIdx.x;
  const int d0   = tid * 2;

  if (tid == 0) {
    __builtin_prefetch(p.w_in, 0, 0);
    __builtin_prefetch(p.w_x, 0, 0);
    __builtin_prefetch(p.w_out, 0, 0);
    __builtin_prefetch(p.w_fc1, 0, 0);
    __builtin_prefetch(p.w_fc2, 0, 0);
  }
  // zero conv history once -> no it==0 special case in the conv loop
  for (int i = tid; i < 3 * 256; i += 128) s_tail[i] = (_Float16)0.f;

  // per-thread channel-local parameters (2 channels per thread)
  float cw[2][4], cb[2], dw[2][4], db[2], Areg[2][16], Dpr[2], hst[2][16];
#pragma unroll
  for (int j = 0; j < 2; ++j) {
    int d = d0 + j;
    cb[j]  = p.conv_b[d];
    db[j]  = p.dt_b[d];
    Dpr[j] = p.Dp[d];
#pragma unroll
    for (int k = 0; k < 4; ++k) { cw[j][k] = p.conv_w[d * 4 + k]; dw[j][k] = p.dt_w[d * 4 + k]; }
#pragma unroll
    for (int n = 0; n < 16; ++n) { Areg[j][n] = -__expf(p.A_log[d * 16 + n]); hst[j][n] = 0.f; }
  }
  if (tid < 64) { s_nw[tid] = p.norm_w[tid]; s_nw2[tid] = p.norm2_w[tid]; }
  __syncthreads();

  const int nTiles = L >> 4;
  for (int it = 0; it < nTiles; ++it) {
    // ---- phase 1: load residual tile (flip rows for bwd direction)
    {
      const float* gbase = xin + (size_t)seq * seqStride;
#if __has_builtin(__builtin_amdgcn_global_load_async_to_lds_b32)
      for (int i = tid; i < 16 * 64; i += 128) {
        int r = i >> 6, c = i & 63;
        int row = it * 16 + r;
        int rin = dir ? (L - 1 - row) : row;
        const float* gp = gbase + rin * rowStride + c;
        __builtin_amdgcn_global_load_async_to_lds_b32(
            (__attribute__((address_space(1))) int*)gp,
            (__attribute__((address_space(3))) int*)&s_xres[i], 0, 0);
      }
      asm volatile("s_wait_asynccnt 0x0" ::: "memory");
#else
      for (int i = tid; i < 16 * 64; i += 128) {
        int r = i >> 6, c = i & 63;
        int row = it * 16 + r;
        int rin = dir ? (L - 1 - row) : row;
        s_xres[i] = gbase[rin * rowStride + c];
      }
#endif
    }
    __syncthreads();

    // ---- phase 2: RMSNorm -> f16 (8 threads per row, shuffle reduce)
    {
      int row = tid >> 3, q = tid & 7;
      float ss = 0.f;
#pragma unroll
      for (int k = 0; k < 8; ++k) { float v = s_xres[row * 64 + q * 8 + k]; ss += v * v; }
      ss += __shfl_xor(ss, 1);
      ss += __shfl_xor(ss, 2);
      ss += __shfl_xor(ss, 4);
      float sc = rsqrtf(ss * (1.f / 64.f) + 1e-5f);
#pragma unroll
      for (int k = 0; k < 8; ++k) {
        int c = q * 8 + k;
        s_hn[row * 64 + c] = (_Float16)(s_xres[row * 64 + c] * sc * s_nw[c]);
      }
    }
    __syncthreads();

    // ---- phase 3: in_proj  (16x64)x(64x512) -> xs|z  (branchless store)
    for (int nt = wave; nt < 32; nt += 4) {
      v8f acc = zero8();
#pragma unroll
      for (int kt = 0; kt < 2; ++kt)
        acc = wmma16(load_a(s_hn, 64, kt * 32, lane), load_b_pk(p.w_in, kt * 32 + nt, lane), acc);
      int n0 = nt * 16 + (lane & 15), mh = (lane >> 4) * 8;
#pragma unroll
      for (int r = 0; r < 8; ++r) s_r0[(mh + r) * 512 + n0] = (_Float16)acc[r];
    }
    __syncthreads();

    // ---- phase 4: causal depthwise conv (k=4) + SiLU -> u
    //      rows 0..2 mix with zero-initialized history (fully unrolled, no branches);
    //      rows 3..15 read only current-tile xs (branch-free steady state)
#pragma unroll
    for (int j = 0; j < 2; ++j) {
      int d = d0 + j;
#pragma unroll
      for (int r = 0; r < 3; ++r) {
        float acc = cb[j];
#pragma unroll
        for (int k = 0; k < 4; ++k) {
          int rr = r - 3 + k; // compile-time constant sign
          float xv = (rr >= 0) ? (float)s_r0[rr * 512 + d]
                               : (float)s_tail[(3 + rr) * 256 + d];
          acc += cw[j][k] * xv;
        }
        s_u[r * 256 + d] = (_Float16)silu_f(acc);
      }
      for (int r = 3; r < 16; ++r) {
        float acc = cb[j];
#pragma unroll
        for (int k = 0; k < 4; ++k)
          acc += cw[j][k] * (float)s_r0[(r - 3 + k) * 512 + d];
        s_u[r * 256 + d] = (_Float16)silu_f(acc);
      }
    }
    __syncthreads();
    // save conv history (xs rows 13..15) before s_r0 gets reused
    for (int i = tid; i < 3 * 256; i += 128)
      s_tail[i] = s_r0[(13 + (i >> 8)) * 512 + (i & 255)];
    __syncthreads();

    // ---- phase 5: x_proj (16x256)x(256x36->48)
    if (wave < 3) {
      int nt = wave;
      v8f acc = zero8();
#pragma unroll
      for (int kt = 0; kt < 8; ++kt)
        acc = wmma16(load_a(s_u, 256, kt * 32, lane), load_b_pk(p.w_x, kt * 3 + nt, lane), acc);
      int n0 = nt * 16 + (lane & 15), mh = (lane >> 4) * 8;
#pragma unroll
      for (int r = 0; r < 8; ++r) s_xdbl[(mh + r) * 48 + n0] = acc[r];
    }
    __syncthreads();

    // ---- phase 6+7: dt (softplus) + selective scan + gating
    for (int r = 0; r < 16; ++r) {
      float Bt[16], Ct[16];
#pragma unroll
      for (int n = 0; n < 16; ++n) { Bt[n] = s_xdbl[r * 48 + 4 + n]; Ct[n] = s_xdbl[r * 48 + 20 + n]; }
#pragma unroll
      for (int j = 0; j < 2; ++j) {
        int d = d0 + j;
        float dtv = db[j];
#pragma unroll
        for (int q = 0; q < 4; ++q) dtv += dw[j][q] * s_xdbl[r * 48 + q];
        dtv = softplus_f(dtv);
        float uv = (float)s_u[r * 256 + d];
        float du = dtv * uv;
        float y  = uv * Dpr[j];
#pragma unroll
        for (int n = 0; n < 16; ++n) {
          hst[j][n] = hst[j][n] * __expf(dtv * Areg[j][n]) + du * Bt[n];
          y += hst[j][n] * Ct[n];
        }
        float zv = (float)s_r0[r * 512 + 256 + d];
        s_y[r * 256 + d] = (_Float16)(y * silu_f(zv));
      }
    }
    __syncthreads();

    // ---- phase 8: out_proj (16x256)x(256x64) + residual -> residual2 in s_xres
    {
      int nt = wave;
      v8f acc = zero8();
#pragma unroll
      for (int kt = 0; kt < 8; ++kt)
        acc = wmma16(load_a(s_y, 256, kt * 32, lane), load_b_pk(p.w_out, kt * 4 + nt, lane), acc);
      int n0 = nt * 16 + (lane & 15), mh = (lane >> 4) * 8;
#pragma unroll
      for (int r = 0; r < 8; ++r) s_xres[(mh + r) * 64 + n0] += acc[r];
    }
    __syncthreads();

    // ---- phase 9: RMSNorm2 -> s_hn (8 threads per row, shuffle reduce)
    {
      int row = tid >> 3, q = tid & 7;
      float ss = 0.f;
#pragma unroll
      for (int k = 0; k < 8; ++k) { float v = s_xres[row * 64 + q * 8 + k]; ss += v * v; }
      ss += __shfl_xor(ss, 1);
      ss += __shfl_xor(ss, 2);
      ss += __shfl_xor(ss, 4);
      float sc = rsqrtf(ss * (1.f / 64.f) + 1e-5f);
#pragma unroll
      for (int k = 0; k < 8; ++k) {
        int c = q * 8 + k;
        s_hn[row * 64 + c] = (_Float16)(s_xres[row * 64 + c] * sc * s_nw2[c]);
      }
    }
    __syncthreads();

    // ---- phase 10: fc1 (16x64)x(64x512) -> y1|gate (f16, reuse s_r0 [16][512])
    for (int nt = wave; nt < 32; nt += 4) {
      v8f acc = zero8();
#pragma unroll
      for (int kt = 0; kt < 2; ++kt)
        acc = wmma16(load_a(s_hn, 64, kt * 32, lane), load_b_pk(p.w_fc1, kt * 32 + nt, lane), acc);
      int n0 = nt * 16 + (lane & 15), mh = (lane >> 4) * 8;
#pragma unroll
      for (int r = 0; r < 8; ++r) s_r0[(mh + r) * 512 + n0] = (_Float16)acc[r];
    }
    __syncthreads();

    // ---- phase 11: gated activation -> s_u (reuse)
#pragma unroll
    for (int j = 0; j < 2; ++j) {
      int d = d0 + j;
      for (int r = 0; r < 16; ++r) {
        float y1 = (float)s_r0[r * 512 + d];
        float g  = (float)s_r0[r * 512 + 256 + d];
        s_u[r * 256 + d] = (_Float16)(y1 * silu_f(g));
      }
    }
    __syncthreads();

    // ---- phase 12: fc2 (16x256)x(256x64) + residual2 -> cat (f16, un-flip rows)
    {
      int nt = wave;
      v8f acc = zero8();
#pragma unroll
      for (int kt = 0; kt < 8; ++kt)
        acc = wmma16(load_a(s_u, 256, kt * 32, lane), load_b_pk(p.w_fc2, kt * 4 + nt, lane), acc);
      int n0 = nt * 16 + (lane & 15), mh = (lane >> 4) * 8;
#pragma unroll
      for (int r = 0; r < 8; ++r) {
        int row  = mh + r;
        float v  = acc[r] + s_xres[row * 64 + n0];
        int rowg = it * 16 + row;
        int orig = dir ? (L - 1 - rowg) : rowg;
        cat[((size_t)(seq * L + orig)) * 128 + dir * 64 + n0] = (_Float16)v;
      }
    }
    __syncthreads();
  }
}

// --------------------------------------------- combine: cat @ W + b + res
// token = tile*16 + m ; seq = token / L ; row = token % L
__global__ void __launch_bounds__(64)
combine_kernel(const _Float16* __restrict__ cat, const _Float16* __restrict__ wpk,
               const float* __restrict__ bias, const float* __restrict__ res,
               float* __restrict__ out, int L,
               int resRowS, int resSeqS, int outRowS, int outSeqS, int outColS) {
  const int tile = blockIdx.x;
  const int lane = threadIdx.x & 31;
  const int wave = threadIdx.x >> 5;
  for (int nt = wave; nt < 4; nt += 2) {
    v8f acc = zero8();
#pragma unroll
    for (int kt = 0; kt < 4; ++kt) {
      v16h a = load_a(cat + (size_t)tile * 16 * 128, 128, kt * 32, lane);
      acc = wmma16(a, load_b_pk(wpk, kt * 4 + nt, lane), acc);
    }
    int n0 = nt * 16 + (lane & 15), mh = (lane >> 4) * 8;
#pragma unroll
    for (int r = 0; r < 8; ++r) {
      int token = tile * 16 + mh + r;
      int seq = token / L, row = token % L;
      float v = acc[r] + bias[n0] + res[row * resRowS + seq * resSeqS + n0];
      out[row * outRowS + seq * outSeqS + n0 * outColS] = v;
    }
  }
}

// ------------------------------------------------------------------ host
extern "C" void kernel_launch(void* const* d_in, const int* in_sizes, int n_in,
                              void* d_out, int out_size, void* d_ws, size_t ws_size,
                              hipStream_t stream) {
  (void)in_sizes; (void)n_in; (void)out_size; (void)ws_size;
  const float* x = (const float*)d_in[0];

  // ---- carve workspace
  char* cur = (char*)d_ws;
  auto carve = [&](size_t bytes) -> void* {
    void* p = cur; cur += (bytes + 255) & ~(size_t)255; return p;
  };
  float*     xt   = (float*)carve((size_t)64 * 256 * 64 * 4);
  float*     xb2  = (float*)carve((size_t)64 * 256 * 64 * 4);
  _Float16*  cat  = (_Float16*)carve((size_t)16384 * 128 * 2);
  _Float16*  pkTl = (_Float16*)carve((size_t)4 * 4 * 512 * 2);
  _Float16*  pkFl = (_Float16*)carve((size_t)4 * 4 * 512 * 2);

  auto packW = [&](const void* src, _Float16* dst, int KT, int NT, int K, int N, int tr) {
    int total = KT * NT * 512;
    pack_weights<<<(total + 255) / 256, 256, 0, stream>>>((const float*)src, dst, KT, NT, K, N, tr);
  };

  // ---- per-block params: d_in flattening (recursive dict-insertion order)
  // block layout (13 arrays): norm_w, in_proj_w, conv_w, conv_b, x_proj_w,
  //   dt_proj_w, dt_proj_b, A_log, D, out_proj_w, norm2_w, fc1_w, fc2_w
  const int bases[4] = {1, 14, 27, 40}; // time.fwd, time.bwd, freq.fwd, freq.bwd
  BlockParams bp[4];
  for (int bi = 0; bi < 4; ++bi) {
    int b = bases[bi];
    _Float16* w_in  = (_Float16*)carve((size_t)2 * 32 * 512 * 2);
    _Float16* w_x   = (_Float16*)carve((size_t)8 * 3  * 512 * 2);
    _Float16* w_out = (_Float16*)carve((size_t)8 * 4  * 512 * 2);
    _Float16* w_fc1 = (_Float16*)carve((size_t)2 * 32 * 512 * 2);
    _Float16* w_fc2 = (_Float16*)carve((size_t)8 * 4  * 512 * 2);
    packW(d_in[b + 1],  w_in,  2, 32, 64,  512, 1);
    packW(d_in[b + 4],  w_x,   8, 3,  256, 36,  1);
    packW(d_in[b + 9],  w_out, 8, 4,  256, 64,  1);
    packW(d_in[b + 11], w_fc1, 2, 32, 64,  512, 1);
    packW(d_in[b + 12], w_fc2, 8, 4,  256, 64,  1);
    bp[bi].norm_w  = (const float*)d_in[b + 0];
    bp[bi].conv_w  = (const float*)d_in[b + 2];
    bp[bi].conv_b  = (const float*)d_in[b + 3];
    bp[bi].dt_w    = (const float*)d_in[b + 5];
    bp[bi].dt_b    = (const float*)d_in[b + 6];
    bp[bi].A_log   = (const float*)d_in[b + 7];
    bp[bi].Dp      = (const float*)d_in[b + 8];
    bp[bi].norm2_w = (const float*)d_in[b + 10];
    bp[bi].w_in = w_in; bp[bi].w_x = w_x; bp[bi].w_out = w_out;
    bp[bi].w_fc1 = w_fc1; bp[bi].w_fc2 = w_fc2;
  }
  packW(d_in[53], pkTl, 4, 4, 128, 64, 0); // tlin_w: y @ W (no transpose)
  packW(d_in[55], pkFl, 4, 4, 128, 64, 0); // flin_w
  const float* tlin_b = (const float*)d_in[54];
  const float* flin_b = (const float*)d_in[56];

  // ---- input transpose: x (1,64,256,64) -> xt[f][t][c]
  transpose_in<<<(64 * 256 * 64 + 255) / 256, 256, 0, stream>>>(x, xt);

  // ---- stage A: time-axis bimamba (seq=f, L=256; addr = f*16384 + t*64 + c)
  mamba_block_kernel<<<64, 128, 0, stream>>>(xt, 256, 64, 16384, 0, bp[0], cat);
  mamba_block_kernel<<<64, 128, 0, stream>>>(xt, 256, 64, 16384, 1, bp[1], cat);
  // tlin + residual -> xb2[f][t][c]   (token = f*256+t; seq=f, row=t)
  combine_kernel<<<1024, 64, 0, stream>>>(cat, pkTl, tlin_b, xt, xb2, 256,
                                          /*resRowS=*/64, /*resSeqS=*/16384,
                                          /*outRowS=*/64, /*outSeqS=*/16384, /*outColS=*/1);

  // ---- stage B: freq-axis bimamba (seq=t, L=64; addr = f*16384 + t*64 + c)
  mamba_block_kernel<<<256, 128, 0, stream>>>(xb2, 64, 16384, 64, 0, bp[2], cat);
  mamba_block_kernel<<<256, 128, 0, stream>>>(xb2, 64, 16384, 64, 1, bp[3], cat);
  // flin + residual, permuted store: d_out[c][t][f] (token = t*64+f; seq=t, row=f)
  combine_kernel<<<1024, 64, 0, stream>>>(cat, pkFl, flin_b, xb2, (float*)d_out, 64,
                                          /*resRowS=*/16384, /*resSeqS=*/64,
                                          /*outRowS=*/1, /*outSeqS=*/64, /*outColS=*/16384);
}
